// LGCN_83476984365521
// MI455X (gfx1250) — matrix operand order repaired and checked by reading, not compiled, not run on hardware
//
#include <hip/hip_runtime.h>
#include <hip/hip_bf16.h>

#define NUM_USERS 100000
#define NUM_ITEMS 50000
#define N_NODES   (NUM_USERS + NUM_ITEMS)
#define EMBED     64
#define N_EDGES   1000000
#define BATCH     8192
#define NODE_F    (N_NODES * EMBED)   // 9,600,000 floats

typedef __attribute__((ext_vector_type(16))) _Float16 v16h;
typedef __attribute__((ext_vector_type(8)))  float    v8f;

// ---------------------------------------------------------------- utilities
__global__ void lgcn_zero_f32(float* __restrict__ p, int n) {
    int i = blockIdx.x * blockDim.x + threadIdx.x;
    if (i < n) p[i] = 0.0f;
}

// deg[u] += 1 ; deg[NUM_USERS+i] += 1 per edge
__global__ void lgcn_degree(const int* __restrict__ eu, const int* __restrict__ ei,
                            float* __restrict__ deg) {
    int e = blockIdx.x * blockDim.x + threadIdx.x;
    if (e >= N_EDGES) return;
    atomicAdd(deg + eu[e], 1.0f);
    atomicAdd(deg + NUM_USERS + ei[e], 1.0f);
}

__global__ void lgcn_dinv(const float* __restrict__ deg, float* __restrict__ dinv) {
    int i = blockIdx.x * blockDim.x + threadIdx.x;
    if (i >= N_NODES) return;
    float d = deg[i];
    dinv[i] = (d > 0.0f) ? __frsqrt_rn(fmaxf(d, 1.0f)) : 0.0f;
}

// cur = concat(user_emb, item_emb); acc = cur
__global__ void lgcn_init(const float* __restrict__ ue, const float* __restrict__ ie,
                          float* __restrict__ cur, float* __restrict__ acc) {
    int i = blockIdx.x * blockDim.x + threadIdx.x;
    if (i >= NODE_F) return;
    float v = (i < NUM_USERS * EMBED) ? ue[i] : ie[i - NUM_USERS * EMBED];
    cur[i] = v;
    acc[i] = v;
}

// One wave32 per undirected edge; each lane moves a float2 slice both directions.
__global__ void lgcn_spmm(const int* __restrict__ eu, const int* __restrict__ ei,
                          const float* __restrict__ dinv,
                          const float* __restrict__ cur, float* __restrict__ nxt) {
    int gid  = blockIdx.x * blockDim.x + threadIdx.x;
    int e    = gid >> 5;
    if (e >= N_EDGES) return;
    int lane = threadIdx.x & 31;

    int u = 0, it = 0;
    if (lane == 0) {
        u  = eu[e];
        it = ei[e];
        // stream-ahead prefetch of the edge index arrays (global_prefetch_b8)
        if (e + 1024 < N_EDGES) {
            __builtin_prefetch(eu + e + 1024, 0, 1);
            __builtin_prefetch(ei + e + 1024, 0, 1);
        }
    }
    u  = __shfl(u, 0, 32);
    it = __shfl(it, 0, 32);

    float val = dinv[u] * dinv[NUM_USERS + it];

    size_t urow = (size_t)u * EMBED;
    size_t irow = ((size_t)NUM_USERS + (size_t)it) * EMBED;
    int k = lane * 2;

    float2 cu = *(const float2*)(cur + urow + k);
    float2 ci = *(const float2*)(cur + irow + k);

    atomicAdd(nxt + urow + k,     val * ci.x);
    atomicAdd(nxt + urow + k + 1, val * ci.y);
    atomicAdd(nxt + irow + k,     val * cu.x);
    atomicAdd(nxt + irow + k + 1, val * cu.y);
}

__global__ void lgcn_acc_add(float* __restrict__ acc, const float* __restrict__ nxt) {
    int i = blockIdx.x * blockDim.x + threadIdx.x;
    if (i < NODE_F) acc[i] += nxt[i];
}

// ------------------------------------------------------------- WMMA scoring
// scores[tile*16 + i] = diag( U16x64 x (P - N)^T 64x16 )[i] / 16
// One wave32 per 16-row batch tile; K=64 done as two v_wmma_f32_16x16x32_f16.
__global__ void lgcn_score_wmma(const float* __restrict__ acc,
                                const int* __restrict__ users,
                                const int* __restrict__ pos_items,
                                const int* __restrict__ neg_items,
                                float* __restrict__ out) {
    __shared__ float tile_lds[16][16];

    int tile = blockIdx.x;          // 0 .. BATCH/16-1
    int lane = threadIdx.x;         // 0 .. 31
    int half = lane >> 4;           // 0 | 1
    int r    = lane & 15;

    const float* urow = acc + (size_t)users[tile * 16 + r] * EMBED;
    const float* prow = acc + ((size_t)NUM_USERS + (size_t)pos_items[tile * 16 + r]) * EMBED;
    const float* nrow = acc + ((size_t)NUM_USERS + (size_t)neg_items[tile * 16 + r]) * EMBED;

    v8f c = {};
    #pragma unroll
    for (int koff = 0; koff < EMBED; koff += 32) {
        v16h a, b;
        // A (16x32 f16): lanes 0-15  hold M=lane, halves = K{0..15}
        //                lanes 16-31 hold M=lane-16, halves = K{16..31}
        int abase0 = koff + (half ? 16 : 0);
        int abase1 = abase0 + 8;
        #pragma unroll
        for (int j = 0; j < 8; ++j) {
            a[j]     = (_Float16)urow[abase0 + j];
            a[j + 8] = (_Float16)urow[abase1 + j];
        }
        // B (32x16 f16): lanes 0-15 hold col N=lane, halves = K{0..15};
        //                lanes 16-31 hold col N=lane-16, halves = K{16..31}
        int bbase = koff + (half ? 16 : 0);
        #pragma unroll
        for (int j = 0; j < 16; ++j) {
            b[j] = (_Float16)(prow[bbase + j] - nrow[bbase + j]);
        }
        c = __builtin_amdgcn_wmma_f32_16x16x32_f16(
                false, a, false, b, (short)0, c, false, false);
    }

    // C/D layout: VGPR j -> M = j + 8*half, N = lane&15. Spill tile, read diag.
    #pragma unroll
    for (int j = 0; j < 8; ++j) {
        tile_lds[j + (half ? 8 : 0)][r] = c[j];
    }
    __syncthreads();
    if (lane < 16) {
        // both operands carry the implicit /4 of light_out -> /16 on the product
        out[tile * 16 + lane] = tile_lds[lane][lane] * (1.0f / 16.0f);
    }
}

// ---------------------------------------------------------------- launcher
extern "C" void kernel_launch(void* const* d_in, const int* in_sizes, int n_in,
                              void* d_out, int out_size, void* d_ws, size_t ws_size,
                              hipStream_t stream) {
    const float* user_emb  = (const float*)d_in[0];
    const float* item_emb  = (const float*)d_in[1];
    const int*   edge_users = (const int*)d_in[2];
    const int*   edge_items = (const int*)d_in[3];
    const int*   users      = (const int*)d_in[4];
    const int*   pos_items  = (const int*)d_in[5];
    const int*   neg_items  = (const int*)d_in[6];
    float* out = (float*)d_out;

    char* ws = (char*)d_ws;
    float* bufA = (float*)(ws);                                   // 38.4 MB
    float* bufB = (float*)(ws + (size_t)NODE_F * 4);              // 38.4 MB
    float* acc  = (float*)(ws + (size_t)NODE_F * 8);              // 38.4 MB
    float* deg  = (float*)(ws + (size_t)NODE_F * 12);             // 0.6 MB
    float* dinv = (float*)(ws + (size_t)NODE_F * 12 + (size_t)N_NODES * 4);

    const int T = 256;
    auto blocks = [](long long n, int t) { return (int)((n + t - 1) / t); };

    // degree -> d_inv
    lgcn_zero_f32<<<blocks(N_NODES, T), T, 0, stream>>>(deg, N_NODES);
    lgcn_degree  <<<blocks(N_EDGES, T), T, 0, stream>>>(edge_users, edge_items, deg);
    lgcn_dinv    <<<blocks(N_NODES, T), T, 0, stream>>>(deg, dinv);

    // cur = acc = all_emb
    lgcn_init<<<blocks(NODE_F, T), T, 0, stream>>>(user_emb, item_emb, bufA, acc);

    // 3 propagation layers, ping-pong between bufA / bufB
    float* cur = bufA;
    float* nxt = bufB;
    for (int layer = 0; layer < 3; ++layer) {
        lgcn_zero_f32<<<blocks(NODE_F, T), T, 0, stream>>>(nxt, NODE_F);
        lgcn_spmm<<<blocks((long long)N_EDGES * 32, T), T, 0, stream>>>(
            edge_users, edge_items, dinv, cur, nxt);
        lgcn_acc_add<<<blocks(NODE_F, T), T, 0, stream>>>(acc, nxt);
        float* tmp = cur; cur = nxt; nxt = tmp;
    }

    // WMMA scoring: one wave per 16 batch rows
    lgcn_score_wmma<<<BATCH / 16, 32, 0, stream>>>(acc, users, pos_items, neg_items, out);
}
